// SEDformer_18683107737655
// MI455X (gfx1250) — compile-verified, matrix operand order checked by reading, not run on hardware
//
#include <hip/hip_runtime.h>
#include <hip/hip_bf16.h>
#include <math.h>

#define LLEN 2048
#define BSZ  16
#define DM   512
#define NH   8
#define DHD  64
#define RF   48
#define HIDN 2048
#define MROWS (LLEN*BSZ)

typedef __attribute__((ext_vector_type(16))) __bf16 v16bf;
typedef __attribute__((ext_vector_type(8)))  __bf16 v8bf;
typedef __attribute__((ext_vector_type(8)))  float  v8f;

// ---------------------------------------------------------------------------
// Async copy of one 16B chunk: global -> LDS, tracked by ASYNCcnt.
// (inline asm: portable across ROCm7.2 / amdgpu-toolchain builtin arity skew)
// ---------------------------------------------------------------------------
__device__ __forceinline__ void async_b128(unsigned lds_byte_addr, const void* gptr)
{
    asm volatile("global_load_async_to_lds_b128 %0, %1, off"
                 :: "v"(lds_byte_addr), "v"(gptr) : "memory");
}

// ---------------------------------------------------------------------------
// Generic bf16 WMMA GEMM:  C[M,N] = A[M,K] (row-major bf16) * Bt[N,K]^T.
// Block = 128 threads = 4 waves; block tile 64x64; K-step 32.
// A/B tiles double-buffered in LDS via GLOBAL_LOAD_ASYNC_TO_LDS_B128
// (ASYNCcnt), WMMA fragments fed by ds_load_b128.
// LDS buffer: [A 64x32 bf16 (4KB) | B 64x32 bf16 (4KB)] x 2 = 16KB dynamic.
// Epilogue: optional bias, BatchNorm, exact GELU, residual; writes f32/bf16.
// ---------------------------------------------------------------------------
__global__ __launch_bounds__(128) void gemm_bf16_ep(
    const __bf16* __restrict__ A, const __bf16* __restrict__ Bt,
    int Mr, int Nc, int Kd,
    const float* __restrict__ bias,
    const float* __restrict__ bng, const float* __restrict__ bnb,
    const float* __restrict__ bnm, const float* __restrict__ bnv,
    const float* __restrict__ resid, int do_gelu,
    float* __restrict__ out32, __bf16* __restrict__ out16)
{
    extern __shared__ char smem[];                       // 16384 bytes
    const unsigned ldsbase = __builtin_amdgcn_groupstaticsize(); // dyn-LDS offset

    const int tid  = threadIdx.x;
    const int lane = tid & 31;
    const int wave = tid >> 5;
    const int m0 = blockIdx.y * 64 + wave * 16;
    const int n0 = blockIdx.x * 64;
    const int rsel = lane >> 4;    // 0: K 0..7 & 16..23, 1: K 8..15 & 24..31
    const int rlo  = lane & 15;
    const int mblk = blockIdx.y * 64;

    // Stage one K-slab (A 64x32 + B 64x32) into LDS buffer `buf`.
    // 512 16B chunks, 4 per thread: c = tid + 128*j; c<256 -> A, else B.
    auto stage = [&](int buf, int k0) {
#pragma unroll
        for (int j = 0; j < 4; ++j) {
            const int c   = tid + 128 * j;
            const int mat = c >> 8;            // 0 = A, 1 = B
            const int idx = c & 255;
            const int row = idx >> 2;
            const int ko  = (idx & 3) * 8;     // element offset within 32
            const __bf16* g = mat
                ? (Bt + (size_t)(n0   + row) * Kd + k0 + ko)
                : (A  + (size_t)(mblk + row) * Kd + k0 + ko);
            const unsigned la = ldsbase + (unsigned)(buf * 8192 + mat * 4096
                                                     + row * 64 + ko * 2);
            async_b128(la, g);
        }
    };

    v8f acc[4];
#pragma unroll
    for (int t = 0; t < 4; ++t)
#pragma unroll
        for (int i = 0; i < 8; ++i) acc[t][i] = 0.0f;

    const int nk = Kd >> 5;
    stage(0, 0);
    for (int s = 0; s < nk; ++s) {
        if (s + 1 < nk) {
            stage((s + 1) & 1, (s + 1) * 32);
            asm volatile("s_wait_asynccnt 0x4" ::: "memory");  // stage s landed
        } else {
            asm volatile("s_wait_asynccnt 0x0" ::: "memory");
        }
        __syncthreads();

        const char* sb = smem + (s & 1) * 8192;
        const char* ar = sb + (wave * 16 + rlo) * 64 + rsel * 16;
        v8bf alo = *(const v8bf*)(ar);
        v8bf ahi = *(const v8bf*)(ar + 32);
        v16bf af;
#pragma unroll
        for (int i = 0; i < 8; ++i) { af[i] = alo[i]; af[i + 8] = ahi[i]; }
#pragma unroll
        for (int t = 0; t < 4; ++t) {
            const char* br = sb + 4096 + (t * 16 + rlo) * 64 + rsel * 16;
            v8bf blo = *(const v8bf*)(br);
            v8bf bhi = *(const v8bf*)(br + 32);
            v16bf bfr;
#pragma unroll
            for (int i = 0; i < 8; ++i) { bfr[i] = blo[i]; bfr[i + 8] = bhi[i]; }
            acc[t] = __builtin_amdgcn_wmma_f32_16x16x32_bf16(
                false, af, false, bfr, (short)0, acc[t], false, false);
        }
        __syncthreads();   // all waves done reading before buffer reuse
    }

    const int rowoff = rsel * 8;   // C layout: VGPR i -> M=i (lanes<16) / i+8
#pragma unroll
    for (int t = 0; t < 4; ++t) {
        const int col = n0 + t * 16 + rlo;
        float sc = 0.f, mc = 0.f, bc = 0.f, bsc = 0.f;
        if (bng) { mc = bnm[col]; bc = bnb[col]; sc = bng[col] * rsqrtf(bnv[col] + 1e-5f); }
        if (bias) bsc = bias[col];
#pragma unroll
        for (int i = 0; i < 8; ++i) {
            const int row = m0 + rowoff + i;
            float val = acc[t][i] + bsc;
            if (bng) val = (val - mc) * sc + bc;
            if (do_gelu) val = 0.5f * val * (1.0f + erff(val * 0.70710678118f));
            const size_t oi = (size_t)row * Nc + col;
            if (resid) val += resid[oi];
            if (out32) out32[oi] = val;
            if (out16) out16[oi] = (__bf16)val;
        }
    }
}

// BN over last dim, f32 in -> bf16 out
__global__ void bn_to_bf16_k(const float* __restrict__ x,
                             const float* __restrict__ g, const float* __restrict__ b,
                             const float* __restrict__ m, const float* __restrict__ v,
                             __bf16* __restrict__ out, int total, int dim)
{
    int i = blockIdx.x * blockDim.x + threadIdx.x;
    if (i >= total) return;
    int c = i % dim;
    float val = (x[i] - m[c]) * (g[c] * rsqrtf(v[c] + 1e-5f)) + b[c];
    out[i] = (__bf16)val;
}

// W[K,N] f32 -> Wt[N,K] bf16
__global__ void transpose_cast_k(const float* __restrict__ W, __bf16* __restrict__ Wt,
                                 int Kd, int Nd)
{
    int i = blockIdx.x * blockDim.x + threadIdx.x;
    if (i >= Kd * Nd) return;
    int n = i / Kd, k = i % Kd;
    Wt[i] = (__bf16)W[(size_t)k * Nd + n];
}

// m_t = beta*m_{t-1} + (1-beta)*x_t  over L, in place; one thread per (b,d)
__global__ void membrane_k(float* __restrict__ data, const float* __restrict__ wtau)
{
    int tid = blockIdx.x * blockDim.x + threadIdx.x;
    if (tid >= BSZ * DM) return;
    int d = tid % DM;
    float w = wtau[d];
    float sp = (w > 20.f) ? w : log1pf(expf(w));     // softplus
    float beta = 1.f - 1.f / (sp + 1.f);
    beta = fminf(fmaxf(beta, 0.f), 0.9999f);
    float omb = 1.f - beta;
    float mstate = 0.f;
    for (int t = 0; t < LLEN; ++t) {
        size_t idx = (size_t)t * (BSZ * DM) + tid;
        mstate = beta * mstate + omb * data[idx];
        data[idx] = mstate;
    }
}

// phi[b,h,l,r] = (elu(x.W + b)+1+1e-6)/sqrt(R)
__global__ void rff_k(const float* __restrict__ xin, const float* __restrict__ W,
                      const float* __restrict__ bvec, float* __restrict__ phi)
{
    int idx = blockIdx.x * blockDim.x + threadIdx.x;
    if (idx >= BSZ * NH * LLEN * RF) return;
    int r = idx % RF;
    int t = idx / RF;
    int l = t % LLEN;
    int bh = t / LLEN;
    int h = bh % NH;
    int b = bh / NH;
    const float* xrow = xin + ((size_t)l * BSZ + b) * DM + h * DHD;
    const float* wcol = W + (size_t)h * DHD * RF + r;
    float p = bvec[h * RF + r];
#pragma unroll 8
    for (int dh = 0; dh < DHD; ++dh) p += xrow[dh] * wcol[(size_t)dh * RF];
    float e = (p > 0.f) ? p : expm1f(p);
    phi[idx] = (e + 1.000001f) * 0.14433756729740643f;   // 1/sqrt(48)
}

// KV[bh,r,d] = sum_l phi_k * v ;  Ksum[bh,r] = clip(sum_l phi_k, 1e-6)
__global__ __launch_bounds__(256) void kv_ksum_k(const float* __restrict__ phik,
    const float* __restrict__ vin, float* __restrict__ kv, float* __restrict__ ksum)
{
    __shared__ float sphi[32 * RF];
    __shared__ float sv[32 * DHD];
    const int tid = threadIdx.x;
    const int bh = blockIdx.x;
    const int h = bh % NH, b = bh / NH;
    float acc[12];
#pragma unroll
    for (int j = 0; j < 12; ++j) acc[j] = 0.f;
    float ks = 0.f;
    for (int l0 = 0; l0 < LLEN; l0 += 32) {
        for (int idx = tid; idx < 32 * RF; idx += 256) {
            int ll = idx / RF, r = idx % RF;
            sphi[idx] = phik[((size_t)bh * LLEN + l0 + ll) * RF + r];
        }
        for (int idx = tid; idx < 32 * DHD; idx += 256) {
            int ll = idx >> 6, d = idx & 63;
            sv[idx] = vin[((size_t)(l0 + ll) * BSZ + b) * DM + h * DHD + d];
        }
        __syncthreads();
#pragma unroll
        for (int j = 0; j < 12; ++j) {
            int cell = tid + 256 * j;
            int r = cell >> 6, d = cell & 63;
            float a = acc[j];
            for (int ll = 0; ll < 32; ++ll) a += sphi[ll * RF + r] * sv[(ll << 6) + d];
            acc[j] = a;
        }
        if (tid < RF) {
            float s = 0.f;
            for (int ll = 0; ll < 32; ++ll) s += sphi[ll * RF + tid];
            ks += s;
        }
        __syncthreads();
    }
#pragma unroll
    for (int j = 0; j < 12; ++j) kv[(size_t)bh * (RF * DHD) + tid + 256 * j] = acc[j];
    if (tid < RF) ksum[bh * RF + tid] = fmaxf(ks, 1e-6f);
}

// attn[l,b,h*64+d] = (phi_q . KV) / (phi_q . Ksum + 1e-6), written as bf16
__global__ __launch_bounds__(256) void attn_out_k(const float* __restrict__ phiq,
    const float* __restrict__ kv, const float* __restrict__ ksum,
    __bf16* __restrict__ attn)
{
    __shared__ float skv[RF * DHD];
    __shared__ float sks[RF];
    __shared__ float sphi[16 * RF];
    const int tid = threadIdx.x;
    const int bh = blockIdx.x;
    const int h = bh % NH, b = bh / NH;
    const int l0 = blockIdx.y * 16;
    for (int idx = tid; idx < RF * DHD; idx += 256) skv[idx] = kv[(size_t)bh * (RF * DHD) + idx];
    if (tid < RF) sks[tid] = ksum[bh * RF + tid];
    for (int idx = tid; idx < 16 * RF; idx += 256) {
        int ll = idx / RF, r = idx % RF;
        sphi[idx] = phiq[((size_t)bh * LLEN + l0 + ll) * RF + r];
    }
    __syncthreads();
    const int d = tid & 63, lg = tid >> 6;
#pragma unroll
    for (int j = 0; j < 4; ++j) {
        int ll = lg * 4 + j;
        float num = 0.f, den = 0.f;
        for (int r = 0; r < RF; ++r) {
            float p = sphi[ll * RF + r];
            num += p * skv[r * DHD + d];
            den += p * sks[r];
        }
        float a = num / (den + 1e-6f);
        attn[((size_t)(l0 + ll) * BSZ + b) * DM + h * DHD + d] = (__bf16)a;
    }
}

extern "C" void kernel_launch(void* const* d_in, const int* in_sizes, int n_in,
                              void* d_out, int out_size, void* d_ws, size_t ws_size,
                              hipStream_t stream)
{
    const float* x     = (const float*)d_in[0];
    const float* bn1g  = (const float*)d_in[1];
    const float* bn1b  = (const float*)d_in[2];
    const float* bn1m  = (const float*)d_in[3];
    const float* bn1v  = (const float*)d_in[4];
    const float* bnqg  = (const float*)d_in[5];
    const float* bnqb  = (const float*)d_in[6];
    const float* bnqm  = (const float*)d_in[7];
    const float* bnqv  = (const float*)d_in[8];
    const float* bnkg  = (const float*)d_in[9];
    const float* bnkb  = (const float*)d_in[10];
    const float* bnkm  = (const float*)d_in[11];
    const float* bnkv  = (const float*)d_in[12];
    const float* bnvg  = (const float*)d_in[13];
    const float* bnvb  = (const float*)d_in[14];
    const float* bnvm  = (const float*)d_in[15];
    const float* bnvv  = (const float*)d_in[16];
    const float* bn2g  = (const float*)d_in[17];
    const float* bn2b  = (const float*)d_in[18];
    const float* bn2m  = (const float*)d_in[19];
    const float* bn2v  = (const float*)d_in[20];
    const float* bf1g  = (const float*)d_in[21];
    const float* bf1b  = (const float*)d_in[22];
    const float* bf1m  = (const float*)d_in[23];
    const float* bf1v  = (const float*)d_in[24];
    const float* bf2g  = (const float*)d_in[25];
    const float* bf2b  = (const float*)d_in[26];
    const float* bf2m  = (const float*)d_in[27];
    const float* bf2v  = (const float*)d_in[28];
    const float* Wq    = (const float*)d_in[29];
    const float* Wk    = (const float*)d_in[30];
    const float* Wv    = (const float*)d_in[31];
    const float* bv    = (const float*)d_in[32];
    const float* Wo    = (const float*)d_in[33];
    const float* bo    = (const float*)d_in[34];
    const float* wtauq = (const float*)d_in[35];
    const float* wtauk = (const float*)d_in[36];
    const float* rffW  = (const float*)d_in[37];
    const float* rffb  = (const float*)d_in[38];
    const float* fc1W  = (const float*)d_in[39];
    const float* fc1b  = (const float*)d_in[40];
    const float* fc2W  = (const float*)d_in[41];
    const float* fc2b  = (const float*)d_in[42];
    float* out = (float*)d_out;

    (void)in_sizes; (void)n_in; (void)out_size; (void)ws_size;

    // ---- workspace layout ----
    char* ws = (char*)d_ws;
    size_t off = 0;
    auto alloc = [&](size_t bytes) -> char* {
        char* p = ws + off;
        off = (off + bytes + 255) & ~(size_t)255;
        return p;
    };
    __bf16* wt_q  = (__bf16*)alloc((size_t)DM * DM * 2);
    __bf16* wt_k  = (__bf16*)alloc((size_t)DM * DM * 2);
    __bf16* wt_v  = (__bf16*)alloc((size_t)DM * DM * 2);
    __bf16* wt_o  = (__bf16*)alloc((size_t)DM * DM * 2);
    __bf16* wt_f1 = (__bf16*)alloc((size_t)DM * HIDN * 2);
    __bf16* wt_f2 = (__bf16*)alloc((size_t)HIDN * DM * 2);
    __bf16* h_bf  = (__bf16*)alloc((size_t)MROWS * DM * 2);   // h1, then h2
    float*  qin   = (float*)alloc((size_t)MROWS * DM * 4);
    float*  kin   = (float*)alloc((size_t)MROWS * DM * 4);
    float*  vin   = (float*)alloc((size_t)MROWS * DM * 4);
    float*  phiq  = (float*)alloc((size_t)BSZ * NH * LLEN * RF * 4);
    float*  phik  = (float*)alloc((size_t)BSZ * NH * LLEN * RF * 4);
    float*  kvbuf = (float*)alloc((size_t)BSZ * NH * RF * DHD * 4);
    float*  ksum  = (float*)alloc((size_t)BSZ * NH * RF * 4);
    __bf16* attnb = (__bf16*)alloc((size_t)MROWS * DM * 2);
    float*  f_x2  = qin;   // reuse: Q dead after phi_q

    const int TB = 256;
    const int SMEM = 16384;   // 2 x (A 4KB + B 4KB) double-buffered tiles

    // weights -> bf16 [N,K]
    transpose_cast_k<<<(DM*DM + TB-1)/TB, TB, 0, stream>>>(Wq, wt_q, DM, DM);
    transpose_cast_k<<<(DM*DM + TB-1)/TB, TB, 0, stream>>>(Wk, wt_k, DM, DM);
    transpose_cast_k<<<(DM*DM + TB-1)/TB, TB, 0, stream>>>(Wv, wt_v, DM, DM);
    transpose_cast_k<<<(DM*DM + TB-1)/TB, TB, 0, stream>>>(Wo, wt_o, DM, DM);
    transpose_cast_k<<<(DM*HIDN + TB-1)/TB, TB, 0, stream>>>(fc1W, wt_f1, DM, HIDN);
    transpose_cast_k<<<(HIDN*DM + TB-1)/TB, TB, 0, stream>>>(fc2W, wt_f2, HIDN, DM);

    // h1 = BN1(x) -> bf16
    bn_to_bf16_k<<<(MROWS*DM + TB-1)/TB, TB, 0, stream>>>(x, bn1g, bn1b, bn1m, bn1v,
                                                          h_bf, MROWS*DM, DM);
    // q/k/v GEMMs + BN (+bias for v)
    dim3 gqkv(DM/64, MROWS/64);
    gemm_bf16_ep<<<gqkv, 128, SMEM, stream>>>(h_bf, wt_q, MROWS, DM, DM,
        nullptr, bnqg, bnqb, bnqm, bnqv, nullptr, 0, qin, nullptr);
    gemm_bf16_ep<<<gqkv, 128, SMEM, stream>>>(h_bf, wt_k, MROWS, DM, DM,
        nullptr, bnkg, bnkb, bnkm, bnkv, nullptr, 0, kin, nullptr);
    gemm_bf16_ep<<<gqkv, 128, SMEM, stream>>>(h_bf, wt_v, MROWS, DM, DM,
        bv, bnvg, bnvb, bnvm, bnvv, nullptr, 0, vin, nullptr);

    // membrane scans (in place)
    membrane_k<<<(BSZ*DM + TB-1)/TB, TB, 0, stream>>>(qin, wtauq);
    membrane_k<<<(BSZ*DM + TB-1)/TB, TB, 0, stream>>>(kin, wtauk);

    // random-feature maps
    const int nrff = BSZ * NH * LLEN * RF;
    rff_k<<<(nrff + TB-1)/TB, TB, 0, stream>>>(qin, rffW, rffb, phiq);
    rff_k<<<(nrff + TB-1)/TB, TB, 0, stream>>>(kin, rffW, rffb, phik);

    // KV / Ksum, then normalized attention output
    kv_ksum_k<<<BSZ*NH, 256, 0, stream>>>(phik, vin, kvbuf, ksum);
    attn_out_k<<<dim3(BSZ*NH, LLEN/16), 256, 0, stream>>>(phiq, kvbuf, ksum, attnb);

    // x2 = x + attn @ Wo + bo
    gemm_bf16_ep<<<gqkv, 128, SMEM, stream>>>(attnb, wt_o, MROWS, DM, DM,
        bo, nullptr, nullptr, nullptr, nullptr, x, 0, f_x2, nullptr);

    // h2 = BN2(x2) -> bf16 (reuse h_bf)
    bn_to_bf16_k<<<(MROWS*DM + TB-1)/TB, TB, 0, stream>>>(f_x2, bn2g, bn2b, bn2m, bn2v,
                                                          h_bf, MROWS*DM, DM);
    // f = gelu(BNf1(h2 @ fc1 + b)) -> bf16
    __bf16* f_bf = (__bf16*)alloc((size_t)MROWS * HIDN * 2);
    gemm_bf16_ep<<<dim3(HIDN/64, MROWS/64), 128, SMEM, stream>>>(h_bf, wt_f1, MROWS, HIDN, DM,
        fc1b, bf1g, bf1b, bf1m, bf1v, nullptr, 1, nullptr, f_bf);
    // out = x2 + BNf2(f @ fc2 + b)
    gemm_bf16_ep<<<dim3(DM/64, MROWS/64), 128, SMEM, stream>>>(f_bf, wt_f2, MROWS, DM, HIDN,
        fc2b, bf2g, bf2b, bf2m, bf2v, f_x2, 0, out, nullptr);
}
